// GraphAttentionLayer_1065151889820
// MI455X (gfx1250) — matrix operand order repaired and checked by reading, not compiled, not run on hardware
//
#include <hip/hip_runtime.h>
#include <hip/hip_bf16.h>
#include <math.h>

// ---------------------------------------------------------------------------
// GraphAttentionLayer on MI455X (gfx1250), wave32 + v_wmma_f32_16x16x32_bf16
// GEMM inner loop: 2x-unrolled ping-pong pipeline (no buffer-rotation movs;
// loads for step s+2 issue while step s's fragments feed the matrix pipe).
// ---------------------------------------------------------------------------

typedef __attribute__((ext_vector_type(16))) __bf16 v16bf;
typedef __attribute__((ext_vector_type(8)))  float  v8f;
typedef __attribute__((ext_vector_type(8)))  unsigned int v8u;

union BF16Frag {
    v8u   u;
    v16bf b;
};

__device__ __forceinline__ unsigned short cvt_bf16_bits(float f) {
    __bf16 h = (__bf16)f;                       // native cvt if HW has it
    return __builtin_bit_cast(unsigned short, h);
}

// ---------------------------------------------------------------------------
// Generic 32x32 tiled transpose with f32 -> bf16 conversion.
//   src: [rows][cols] f32, dst: [cols][rows] bf16
// blockDim = (32, 8); grid = (cols/32, rows/32, batches)
// ---------------------------------------------------------------------------
__global__ __launch_bounds__(256) void transpose_cvt_bf16(
    const float* __restrict__ src, unsigned short* __restrict__ dst,
    int rows, int cols, long sbatch_src, long sbatch_dst)
{
    __shared__ float tile[32][33];
    const float* s = src + (long)blockIdx.z * sbatch_src;
    unsigned short* d = dst + (long)blockIdx.z * sbatch_dst;

    int c0 = blockIdx.x * 32;
    int r0 = blockIdx.y * 32;
    int tx = threadIdx.x, ty = threadIdx.y;

    #pragma unroll
    for (int j = 0; j < 4; ++j) {
        int r = r0 + ty + j * 8;
        tile[ty + j * 8][tx] = s[(long)r * cols + c0 + tx];
    }
    __syncthreads();
    #pragma unroll
    for (int j = 0; j < 4; ++j) {
        int c = c0 + ty + j * 8;
        d[(long)c * rows + r0 + tx] = cvt_bf16_bits(tile[tx][ty + j * 8]);
    }
}

// ---------------------------------------------------------------------------
// e[b][n][h] = sum_d Wh[b][n][h*64+d]^2     (65536 values, 64-float sums)
// ---------------------------------------------------------------------------
__global__ __launch_bounds__(256) void head_sqnorms(
    const float* __restrict__ Wh, float* __restrict__ e)
{
    int idx = blockIdx.x * blockDim.x + threadIdx.x;     // [0, 65536)
    int h  = idx & 3;
    int bn = idx >> 2;
    const float4* p = (const float4*)(Wh + (long)bn * 256 + h * 64);
    float s = 0.0f;
    #pragma unroll
    for (int i = 0; i < 16; ++i) {
        float4 v = p[i];
        s += v.x * v.x + v.y * v.y + v.z * v.z + v.w * v.w;
    }
    e[idx] = s;
}

// ---------------------------------------------------------------------------
// Per-(b,h) softmax stats over 2048 nodes: ms[bh] = {max, sum_exp}
// grid = 32 blocks, block = 256
// ---------------------------------------------------------------------------
__global__ __launch_bounds__(256) void softmax_stats(
    const float* __restrict__ e, float* __restrict__ ms)
{
    __shared__ float red[256];
    int bh = blockIdx.x;                 // b*4 + h
    int b = bh >> 2, h = bh & 3;
    const float* ep = e + (long)b * 2048 * 4 + h;
    int tid = threadIdx.x;

    float m = -1e30f;
    for (int n = tid; n < 2048; n += 256) m = fmaxf(m, ep[(long)n * 4]);
    red[tid] = m; __syncthreads();
    for (int s = 128; s > 0; s >>= 1) {
        if (tid < s) red[tid] = fmaxf(red[tid], red[tid + s]);
        __syncthreads();
    }
    m = red[0]; __syncthreads();

    float sum = 0.0f;
    for (int n = tid; n < 2048; n += 256) sum += __expf(ep[(long)n * 4] - m);
    red[tid] = sum; __syncthreads();
    for (int s = 128; s > 0; s >>= 1) {
        if (tid < s) red[tid] += red[tid + s];
        __syncthreads();
    }
    if (tid == 0) { ms[bh * 2] = m; ms[bh * 2 + 1] = red[0]; }
}

// ---------------------------------------------------------------------------
// att[b][n] = mean_h exp(e[b][n][h]-max_bh)/sum_bh      (16384 threads)
// ---------------------------------------------------------------------------
__global__ __launch_bounds__(256) void compute_att(
    const float* __restrict__ e, const float* __restrict__ ms,
    float* __restrict__ att)
{
    int idx = blockIdx.x * blockDim.x + threadIdx.x;     // [0, 16384)
    int b = idx >> 11;
    const float* ep = e + (long)idx * 4;
    float a = 0.0f;
    #pragma unroll
    for (int h = 0; h < 4; ++h) {
        float m = ms[(b * 4 + h) * 2];
        float s = ms[(b * 4 + h) * 2 + 1];
        a += __expf(ep[h] - m) / s;
    }
    att[idx] = 0.25f * a;
}

// ---------------------------------------------------------------------------
// WMMA bf16 GEMM:  C[b] = epilogue( A[b](f32,[M][K]) * Bt[b](bf16,[Ncols][K]) )
// A is converted f32->bf16 in registers.  Bt is K-contiguous per column so
// B-fragments load as 2x uint4 per lane (32x16 B layout: lanes 0-15 K=0-7,16-23,
// lanes 16-31 K=8-15,24-31).
//
// Block = 256 threads = 8 waves: wave (wm 0..1, wn 0..3); block tile = 32 rows
// x 256 cols; each wave: 16 rows x 64 cols = 4 accumulator tiles.
// K must be a multiple of 64 (2x-unrolled ping-pong pipeline).
// Epilogue: x = acc * rowscale[b][m] + bias[n]; optional relu.
// grid = (M/32, batches)
// ---------------------------------------------------------------------------
__global__ __launch_bounds__(256) void gemm_bf16_wmma(
    const float* __restrict__ A,
    const unsigned short* __restrict__ Bt,
    float* __restrict__ C,
    const float* __restrict__ bias,       // may be null
    const float* __restrict__ rowscale,   // may be null; [batch][M]
    int M, int Ncols, int K,
    long sA, long sB, long sC,
    int relu)
{
    int batch = blockIdx.y;
    const float* Ab          = A  + (long)batch * sA;
    const unsigned short* Bb = Bt + (long)batch * sB;
    float* Cb                = C  + (long)batch * sC;
    const float* rs          = rowscale ? rowscale + (long)batch * M : nullptr;

    int tid  = threadIdx.x;
    int lane = tid & 31;
    int wave = tid >> 5;
    int wm = wave & 1;          // 0..1 -> M sub-tile
    int wn = wave >> 1;         // 0..3 -> N quarter
    int m0 = blockIdx.x * 32 + wm * 16;
    int n0 = wn * 64;

    int row  = lane & 15;       // A row within tile / B-D column within tile
    int half = lane >> 4;       // K-half selector

    v8f acc[4] = {};

    const float* Arow = Ab + (long)(m0 + row) * K + half * 8;

    // Per-tile B column base pointers (lane-fixed column, K-contiguous)
    const unsigned short* Bcol[4];
    #pragma unroll
    for (int t = 0; t < 4; ++t)
        Bcol[t] = Bb + (long)(n0 + t * 16 + row) * K + half * 8;

    // ---- fragment loaders -------------------------------------------------
    auto load_a = [&](int k, float4* dst) {
        const float* ap = Arow + k;
        dst[0] = *(const float4*)(ap);
        dst[1] = *(const float4*)(ap + 4);
        dst[2] = *(const float4*)(ap + 16);
        dst[3] = *(const float4*)(ap + 20);
    };
    auto load_b = [&](int k, BF16Frag* dst) {
        #pragma unroll
        for (int t = 0; t < 4; ++t) {
            const unsigned short* bp = Bcol[t] + k;
            uint4 lo = *(const uint4*)(bp);
            uint4 hi = *(const uint4*)(bp + 16);
            dst[t].u[0] = lo.x; dst[t].u[1] = lo.y;
            dst[t].u[2] = lo.z; dst[t].u[3] = lo.w;
            dst[t].u[4] = hi.x; dst[t].u[5] = hi.y;
            dst[t].u[6] = hi.z; dst[t].u[7] = hi.w;
        }
    };
    // Native f32->bf16 conversion; clang vectorizes adjacent pairs into
    // v_cvt_pk_bf16_f32 when the target supports it.
    auto pack_a = [&](const float4* src, BF16Frag& f) {
        v16bf r;
        r[0]  = (__bf16)src[0].x; r[1]  = (__bf16)src[0].y;
        r[2]  = (__bf16)src[0].z; r[3]  = (__bf16)src[0].w;
        r[4]  = (__bf16)src[1].x; r[5]  = (__bf16)src[1].y;
        r[6]  = (__bf16)src[1].z; r[7]  = (__bf16)src[1].w;
        r[8]  = (__bf16)src[2].x; r[9]  = (__bf16)src[2].y;
        r[10] = (__bf16)src[2].z; r[11] = (__bf16)src[2].w;
        r[12] = (__bf16)src[3].x; r[13] = (__bf16)src[3].y;
        r[14] = (__bf16)src[3].z; r[15] = (__bf16)src[3].w;
        f.b = r;
    };
    auto do_wmma = [&](const BF16Frag& af, const BF16Frag* bf) {
        #pragma unroll
        for (int t = 0; t < 4; ++t) {
            acc[t] = __builtin_amdgcn_wmma_f32_16x16x32_bf16(
                /*neg_a=*/false, af.b, /*neg_b=*/false, bf[t].b,
                /*c_mod=*/(short)0, acc[t],
                /*reuse_a=*/false, /*reuse_b=*/false);
        }
    };

    // ---- prologue: stage K-steps 0 and 32 into the two parity sets --------
    float4   araw0[4], araw1[4];
    BF16Frag bfr0[4], bfr1[4];
    load_a(0,  araw0);
    load_b(0,  bfr0);
    load_a(32, araw1);
    load_b(32, bfr1);

    // K % 64 == 0 required
    for (int k0 = 0; k0 < K; k0 += 64) {
        // ---- half A: consume parity-0 set, then re-load it for k0+64 ------
        BF16Frag afrag0;
        pack_a(araw0, afrag0);
        do_wmma(afrag0, bfr0);

        int kn0 = (k0 + 64 < K) ? (k0 + 64) : k0;   // dummy reload at tail
        load_a(kn0, araw0);
        load_b(kn0, bfr0);
        __builtin_prefetch(Arow + kn0 + 64, 0, 1);  // stream adj (global_prefetch_b8)

        // ---- half B: consume parity-1 set, then re-load it for k0+96 ------
        BF16Frag afrag1;
        pack_a(araw1, afrag1);
        do_wmma(afrag1, bfr1);

        int kn1 = (k0 + 96 < K) ? (k0 + 96) : (k0 + 32);
        load_a(kn1, araw1);
        load_b(kn1, bfr1);
    }

    // ---- epilogue: D layout -> VGPR v: lanes0-15 M=v, lanes16-31 M=v+8
    #pragma unroll
    for (int t = 0; t < 4; ++t) {
        int n = n0 + t * 16 + row;
        float bv = bias ? bias[n] : 0.0f;
        #pragma unroll
        for (int v = 0; v < 8; ++v) {
            int m = m0 + half * 8 + v;
            float sc = rs ? rs[m] : 1.0f;
            float x = acc[t][v] * sc + bv;
            if (relu) x = fmaxf(x, 0.0f);
            Cb[(long)m * Ncols + n] = x;
        }
    }
}

// ---------------------------------------------------------------------------
// Host-side launcher
// ---------------------------------------------------------------------------
extern "C" void kernel_launch(void* const* d_in, const int* in_sizes, int n_in,
                              void* d_out, int out_size, void* d_ws, size_t ws_size,
                              hipStream_t stream) {
    (void)in_sizes; (void)n_in; (void)out_size; (void)ws_size;

    const int B = 8, N = 2048, F = 256, HD = 256;   // H*D = 256

    const float* features = (const float*)d_in[0];  // [B][N][F]
    const float* adj      = (const float*)d_in[1];  // [B][N][N]
    const float* kern     = (const float*)d_in[2];  // [F][HD]
    const float* bias     = (const float*)d_in[3];  // [HD]
    float* out            = (float*)d_out;          // [B][N][HD]

    // workspace partition (all offsets 256B aligned)
    char* w = (char*)d_ws;
    float*          Wh  = (float*)(w);                           // 16 MB
    unsigned short* WhT = (unsigned short*)(w + (16l << 20));    //  8 MB
    unsigned short* kT  = (unsigned short*)(w + (24l << 20));    // 128 KB
    float*          e   = (float*)(w + (24l << 20) + 131072);    // 256 KB
    float*          att = (float*)(w + (24l << 20) + 131072 + 262144); // 64 KB
    float*          ms  = (float*)(w + (24l << 20) + 131072 + 262144 + 65536);

    // 1) kT[n][k] = bf16(kernel[k][n])
    transpose_cvt_bf16<<<dim3(HD / 32, F / 32, 1), dim3(32, 8), 0, stream>>>(
        kern, kT, F, HD, 0, 0);

    // 2) Wh = features @ kernel   (M = B*N = 16384, K = F = 256)
    gemm_bf16_wmma<<<dim3((B * N) / 32, 1), 256, 0, stream>>>(
        features, kT, Wh, nullptr, nullptr,
        B * N, HD, F, 0, 0, 0, /*relu=*/0);

    // 3) e[b][n][h] = ||Wh head||^2
    head_sqnorms<<<dim3((B * N * 4) / 256), 256, 0, stream>>>(Wh, e);

    // 4) per-(b,h) softmax stats over nodes
    softmax_stats<<<dim3(B * 4), 256, 0, stream>>>(e, ms);

    // 5) att[b][n] = head-mean of softmax
    compute_att<<<dim3((B * N) / 256), 256, 0, stream>>>(e, ms, att);

    // 6) WhT[b][d][n] = bf16(Wh[b][n][d])
    transpose_cvt_bf16<<<dim3(HD / 32, N / 32, B), dim3(32, 8), 0, stream>>>(
        Wh, WhT, N, HD, (long)N * HD, (long)HD * N);

    // 7) out = relu(att[b][m] * (adj @ Wh) + bias)
    gemm_bf16_wmma<<<dim3(N / 32, B), 256, 0, stream>>>(
        adj, WhT, out, bias, att,
        N, HD, N,
        (long)N * N, (long)HD * N, (long)N * HD, /*relu=*/1);
}
